// BaseTransformerModel_19129784336837
// MI455X (gfx1250) — matrix operand order repaired and checked by reading, not compiled, not run on hardware
//
#include <hip/hip_runtime.h>
#include <hip/hip_bf16.h>
#include <cstdint>

// ---------------------------------------------------------------------------
// GPT forward (B=4,T=1024,D=1024,H=16,HS=64,L=6,V=32000) for gfx1250 (MI455X)
// - all matmuls through v_wmma_f32_16x16x32_bf16 (wave32)
// - GEMM tiles staged with GLOBAL_LOAD_ASYNC_TO_LDS_B128 + double buffering
//   (ASYNCcnt-tracked DMA overlapped with WMMA compute)
// Workspace requirement: ~152 MB.
// ---------------------------------------------------------------------------

typedef unsigned short u16;
typedef __bf16 bf16_t;
typedef __attribute__((ext_vector_type(16))) bf16_t v16bf;
typedef __attribute__((ext_vector_type(8)))  float  v8f;
typedef __attribute__((ext_vector_type(8)))  u16    us8;

union FragAB { us8 h[2]; v16bf v; };

__device__ __forceinline__ u16 f2bf(float f) {
  union { float f; unsigned u; } x; x.f = f;
  unsigned r = x.u + 0x7FFFu + ((x.u >> 16) & 1u);   // round-to-nearest-even
  return (u16)(r >> 16);
}

// --- CDNA5 async global->LDS copy (16B per lane), ASYNCcnt tracked ----------
typedef int v4i32 __attribute__((vector_size(16)));
typedef __attribute__((address_space(1))) v4i32 gv4i;   // global int4
typedef __attribute__((address_space(3))) v4i32 lv4i;   // LDS int4

__device__ __forceinline__ void async_copy_b128(const u16* gsrc, u16* ldst) {
#if __has_builtin(__builtin_amdgcn_global_load_async_to_lds_b128)
  __builtin_amdgcn_global_load_async_to_lds_b128(
      (gv4i*)gsrc, (lv4i*)ldst, 0, 0);
#else
  unsigned l = (unsigned)(uintptr_t)ldst;   // LDS window: low 32 bits = offset
  asm volatile("global_load_async_to_lds_b128 %0, %1, off"
               :: "v"(l), "v"(gsrc) : "memory");
#endif
}

__device__ __forceinline__ void wait_async0() {
#if __has_builtin(__builtin_amdgcn_s_wait_asynccnt)
  __builtin_amdgcn_s_wait_asynccnt(0);
#else
  asm volatile("s_wait_asynccnt 0x0" ::: "memory");
#endif
}

// ---------------------------------------------------------------------------
// Embedding: x[bt,d] = tok_emb[tok[bt],d] + pos_emb[t,d]
// ---------------------------------------------------------------------------
__global__ __launch_bounds__(256) void embed_kernel(
    const int* __restrict__ tok, const float* __restrict__ te,
    const float* __restrict__ pe, float* __restrict__ x) {
  int idx = blockIdx.x * 256 + threadIdx.x;      // 4096*1024 total
  int bt = idx >> 10, d = idx & 1023;
  int t = bt & 1023;                              // T == 1024
  int tk = tok[bt];
  x[idx] = te[(size_t)tk * 1024 + d] + pe[t * 1024 + d];
}

// ---------------------------------------------------------------------------
// LayerNorm over D=1024, writes bf16 (WMMA A-operand ready). 1 block per row.
// ---------------------------------------------------------------------------
__global__ __launch_bounds__(256) void ln_kernel(
    const float* __restrict__ X, const float* __restrict__ g,
    const float* __restrict__ b, u16* __restrict__ out) {
  int row = blockIdx.x;
  const float* xr = X + (size_t)row * 1024;
  float v[4]; float s = 0.f, s2 = 0.f;
#pragma unroll
  for (int i = 0; i < 4; i++) {
    int d = threadIdx.x + i * 256;
    v[i] = xr[d]; s += v[i]; s2 += v[i] * v[i];
  }
#pragma unroll
  for (int m = 1; m < 32; m <<= 1) {
    s  += __shfl_xor(s,  m, 32);
    s2 += __shfl_xor(s2, m, 32);
  }
  __shared__ float a1[8], a2[8];
  int wid = threadIdx.x >> 5, lane = threadIdx.x & 31;
  if (lane == 0) { a1[wid] = s; a2[wid] = s2; }
  __syncthreads();
  float ts = 0.f, ts2 = 0.f;
#pragma unroll
  for (int wv = 0; wv < 8; wv++) { ts += a1[wv]; ts2 += a2[wv]; }
  float mu  = ts  * (1.f / 1024.f);
  float var = ts2 * (1.f / 1024.f) - mu * mu;
  float rs  = rsqrtf(var + 1e-5f);
#pragma unroll
  for (int i = 0; i < 4; i++) {
    int d = threadIdx.x + i * 256;
    out[(size_t)row * 1024 + d] = f2bf((v[i] - mu) * rs * g[d] + b[d]);
  }
}

// ---------------------------------------------------------------------------
// fp32 -> bf16 weight conversion (one-shot per matrix, HBM-cheap)
// ---------------------------------------------------------------------------
__global__ __launch_bounds__(256) void cvt_kernel(
    const float* __restrict__ in, u16* __restrict__ out) {
  int i = blockIdx.x * 256 + threadIdx.x;
  out[i] = f2bf(in[i]);
}

// ---------------------------------------------------------------------------
// Pack per-head weights [H,D,HS] fp32 -> row-major [D, H*HS] bf16
// ---------------------------------------------------------------------------
__global__ __launch_bounds__(256) void pack_qkv_kernel(
    const float* __restrict__ W, u16* __restrict__ out) {
  int idx = blockIdx.x * 256 + threadIdx.x;      // 1024*1024 total
  int d = idx >> 10, c = idx & 1023;
  int h = c >> 6, s = c & 63;
  out[idx] = f2bf(W[((size_t)h * 1024 + d) * 64 + s]);
}

// ---------------------------------------------------------------------------
// GEMM: C[M,N](f32 or bf16) = A[M,K](bf16) * B[K,N](bf16) + bias (+ReLU)
// (+residual). Block tile 128x128, BK=32, 8 waves each owning 32x64 (2x4
// wmma accumulators). Double-buffered LDS; tiles DMA'd via async-to-LDS so
// the next tile streams while WMMA consumes the current one.
// ---------------------------------------------------------------------------
template <bool RELU, bool RESID, bool OUT_BF16>
__global__ __launch_bounds__(256) void gemm_wmma(
    const u16* __restrict__ A, const u16* __restrict__ Bm,
    const float* __restrict__ bias, const float* __restrict__ resid,
    void* __restrict__ Cout, int M, int N, int K) {
  __shared__ __align__(16) u16 sA[2][128 * 32];
  __shared__ __align__(16) u16 sB[2][32 * 128];
  const int tid = threadIdx.x, lane = tid & 31, w = tid >> 5;
  const int wm = w >> 1, wn = w & 1;
  const int m0 = blockIdx.y * 128, n0 = blockIdx.x * 128;
  const int g = (lane < 16) ? 0 : 8;

  v8f acc[2][4];
#pragma unroll
  for (int i = 0; i < 2; i++)
#pragma unroll
    for (int j = 0; j < 4; j++)
#pragma unroll
      for (int e = 0; e < 8; e++) acc[i][j][e] = 0.f;

  // issue one 128x32 A tile + 32x128 B tile (4 async b128 per thread)
  auto stage = [&](int buf, int k0) {
#pragma unroll
    for (int i = 0; i < 2; i++) {                 // A: 512 chunks of 8 bf16
      int c = tid + i * 256;
      int r = c >> 2, cc = (c & 3) * 8;
      async_copy_b128(&A[(size_t)(m0 + r) * K + k0 + cc], &sA[buf][r * 32 + cc]);
    }
#pragma unroll
    for (int i = 0; i < 2; i++) {                 // B: 512 chunks of 8 bf16
      int c = tid + i * 256;
      int r = c >> 4, cc = (c & 15) * 8;
      async_copy_b128(&Bm[(size_t)(k0 + r) * N + n0 + cc], &sB[buf][r * 128 + cc]);
    }
  };

  const int nk = K >> 5;
  stage(0, 0);
  for (int it = 0; it < nk; ++it) {
    wait_async0();          // own async ops for tile(it) complete
    __syncthreads();        // all waves staged tile(it); prev compute done
    if (it + 1 < nk) stage((it + 1) & 1, (it + 1) << 5);
    const int buf = it & 1;

    FragAB af[2], bf[4];
#pragma unroll
    for (int mt = 0; mt < 2; mt++) {
      int r = wm * 32 + mt * 16 + (lane & 15);
      af[mt].h[0] = *(const us8*)&sA[buf][r * 32 + g];
      af[mt].h[1] = *(const us8*)&sA[buf][r * 32 + g + 16];
    }
#pragma unroll
    for (int nt = 0; nt < 4; nt++) {
      int nb = wn * 64 + nt * 16;
#pragma unroll
      for (int e = 0; e < 16; e++)
        bf[nt].h[e >> 3][e & 7] = sB[buf][(g + (e & 7) + (e >> 3) * 16) * 128 + nb + (lane & 15)];
    }
#pragma unroll
    for (int mt = 0; mt < 2; mt++)
#pragma unroll
      for (int nt = 0; nt < 4; nt++)
        acc[mt][nt] = __builtin_amdgcn_wmma_f32_16x16x32_bf16(
            false, af[mt].v, false, bf[nt].v, (short)0, acc[mt][nt], false, false);
  }

  // epilogue: bias / relu / residual, f32 or bf16 store
  const int hi8 = (lane < 16) ? 0 : 8;
#pragma unroll
  for (int mt = 0; mt < 2; mt++) {
#pragma unroll
    for (int nt = 0; nt < 4; nt++) {
      int col = n0 + wn * 64 + nt * 16 + (lane & 15);
      float bv = bias ? bias[col] : 0.f;
#pragma unroll
      for (int j = 0; j < 8; j++) {
        int row = m0 + wm * 32 + mt * 16 + j + hi8;
        float val = acc[mt][nt][j] + bv;
        if (RELU) val = fmaxf(val, 0.f);
        size_t o = (size_t)row * N + col;
        if (RESID) val += resid[o];
        if (OUT_BF16) ((u16*)Cout)[o] = f2bf(val);
        else          ((float*)Cout)[o] = val;
      }
    }
  }
}

// ---------------------------------------------------------------------------
// Flash attention: one block = 128 queries of one (b,h); 8 waves x 16 rows.
// K-chunks of 64 keys staged transposed into LDS; online softmax per wave;
// P transposed C->A layout through wave-private LDS; V fragments from global.
// ---------------------------------------------------------------------------
__global__ __launch_bounds__(256) void attn_kernel(
    const u16* __restrict__ Q, const u16* __restrict__ Kg,
    const u16* __restrict__ V, u16* __restrict__ O) {
  __shared__ __align__(16) u16 sKT[64 * 72];       // [hs][key], padded
  __shared__ __align__(16) u16 sP[8 * 16 * 72];    // per-wave P scratch
  const int tid = threadIdx.x, lane = tid & 31, w = tid >> 5;
  const int bh = blockIdx.y;                       // b*16 + h
  const int bb = bh >> 4, h = bh & 15;
  const int q0 = blockIdx.x * 128;
  const int qbase = q0 + w * 16;
  const int nlow = lane & 15;
  const int g = (lane < 16) ? 0 : 8;
  const int hi8 = (lane < 16) ? 0 : 8;
  const size_t baseBT = (size_t)bb * 1024;

  // Q fragments for this wave's 16 rows (hs 0..63 -> 2 A-frags)
  FragAB qf[2];
  {
    const u16* qrow = Q + (baseBT + qbase + nlow) * 1024 + h * 64;
#pragma unroll
    for (int c = 0; c < 2; c++) {
      qf[c].h[0] = *(const us8*)(qrow + c * 32 + g);
      qf[c].h[1] = *(const us8*)(qrow + c * 32 + g + 16);
    }
  }

  v8f o[4];
#pragma unroll
  for (int ot = 0; ot < 4; ot++)
#pragma unroll
    for (int e = 0; e < 8; e++) o[ot][e] = 0.f;
  float mrow[8], lrow[8];
#pragma unroll
  for (int j = 0; j < 8; j++) { mrow[j] = -1e30f; lrow[j] = 0.f; }

  u16* sPw = &sP[w * 16 * 72];
  const int qmax = q0 + 127;

  for (int kc = 0; kc <= qmax; kc += 64) {
    __syncthreads();
    // stage K^T chunk: sKT[hs][key]  (coalesced global read, transposed write)
#pragma unroll
    for (int i = 0; i < 16; i++) {
      int e = tid + i * 256;
      int key = e >> 6, hs = e & 63;
      sKT[hs * 72 + key] = Kg[(baseBT + kc + key) * 1024 + h * 64 + hs];
    }
    __syncthreads();

    // S = Q K^T  (4 key-tiles x 2 hs-chunks)
    v8f s[4];
#pragma unroll
    for (int nt = 0; nt < 4; nt++) {
#pragma unroll
      for (int e = 0; e < 8; e++) s[nt][e] = 0.f;
#pragma unroll
      for (int hc = 0; hc < 2; hc++) {
        FragAB bfr;
#pragma unroll
        for (int e = 0; e < 16; e++)
          bfr.h[e >> 3][e & 7] = sKT[(hc * 32 + g + (e & 7) + (e >> 3) * 16) * 72 + nt * 16 + nlow];
        s[nt] = __builtin_amdgcn_wmma_f32_16x16x32_bf16(
            false, qf[hc].v, false, bfr.v, (short)0, s[nt], false, false);
      }
    }

    // scale + causal mask
#pragma unroll
    for (int nt = 0; nt < 4; nt++) {
      int key = kc + nt * 16 + nlow;
#pragma unroll
      for (int j = 0; j < 8; j++) {
        int qr = qbase + j + hi8;
        float sv = s[nt][j] * 0.125f;     // HS^-0.5 = 1/8
        s[nt][j] = (key <= qr) ? sv : -1e30f;
      }
    }

    // online softmax (row reduction across the 16-lane half)
#pragma unroll
    for (int j = 0; j < 8; j++) {
      float rm = fmaxf(fmaxf(s[0][j], s[1][j]), fmaxf(s[2][j], s[3][j]));
      rm = fmaxf(rm, __shfl_xor(rm, 1, 32));
      rm = fmaxf(rm, __shfl_xor(rm, 2, 32));
      rm = fmaxf(rm, __shfl_xor(rm, 4, 32));
      rm = fmaxf(rm, __shfl_xor(rm, 8, 32));
      float nm = fmaxf(mrow[j], rm);
      float alpha = __expf(mrow[j] - nm);
      mrow[j] = nm;
      float rsum = 0.f;
#pragma unroll
      for (int nt = 0; nt < 4; nt++) {
        float p = (s[nt][j] > -1e29f) ? __expf(s[nt][j] - nm) : 0.f;
        s[nt][j] = p;
        rsum += p;
      }
      rsum += __shfl_xor(rsum, 1, 32);
      rsum += __shfl_xor(rsum, 2, 32);
      rsum += __shfl_xor(rsum, 4, 32);
      rsum += __shfl_xor(rsum, 8, 32);
      lrow[j] = lrow[j] * alpha + rsum;
#pragma unroll
      for (int ot = 0; ot < 4; ot++) o[ot][j] *= alpha;
    }

    // P -> wave-private LDS (C-layout scatter; A-layout gather below).
    // LDS ops from the same wave are in-order, so no barrier needed.
#pragma unroll
    for (int nt = 0; nt < 4; nt++) {
      int col = nt * 16 + nlow;
#pragma unroll
      for (int j = 0; j < 8; j++) sPw[(j + hi8) * 72 + col] = f2bf(s[nt][j]);
    }

    // O += P V
#pragma unroll
    for (int ksub = 0; ksub < 2; ksub++) {
      FragAB pf;
      pf.h[0] = *(const us8*)&sPw[nlow * 72 + ksub * 32 + g];
      pf.h[1] = *(const us8*)&sPw[nlow * 72 + ksub * 32 + g + 16];
      const u16* vrow = V + (baseBT + kc + ksub * 32) * 1024 + h * 64;
#pragma unroll
      for (int ot = 0; ot < 4; ot++) {
        FragAB vf;
#pragma unroll
        for (int e = 0; e < 16; e++)
          vf.h[e >> 3][e & 7] = vrow[(size_t)(g + (e & 7) + (e >> 3) * 16) * 1024 + ot * 16 + nlow];
        o[ot] = __builtin_amdgcn_wmma_f32_16x16x32_bf16(
            false, pf.v, false, vf.v, (short)0, o[ot], false, false);
      }
    }
  }

  // normalize and store bf16 (concat-heads layout [B,T,H*HS])
#pragma unroll
  for (int j = 0; j < 8; j++) {
    float inv = 1.f / lrow[j];
    int qr = qbase + j + hi8;
#pragma unroll
    for (int ot = 0; ot < 4; ot++)
      O[(baseBT + qr) * 1024 + h * 64 + ot * 16 + nlow] = f2bf(o[ot][j] * inv);
  }
}

// ---------------------------------------------------------------------------
// Host orchestration
// ---------------------------------------------------------------------------
extern "C" void kernel_launch(void* const* d_in, const int* in_sizes, int n_in,
                              void* d_out, int out_size, void* d_ws, size_t ws_size,
                              hipStream_t stream) {
  (void)in_sizes; (void)n_in; (void)out_size; (void)ws_size;
  const int*   tokens  = (const int*)d_in[0];
  const float* tok_emb = (const float*)d_in[1];
  const float* pos_emb = (const float*)d_in[2];
  const float* Wq   = (const float*)d_in[3];
  const float* Wk   = (const float*)d_in[4];
  const float* Wv   = (const float*)d_in[5];
  const float* Wo   = (const float*)d_in[6];
  const float* bo   = (const float*)d_in[7];
  const float* ln1g = (const float*)d_in[8];
  const float* ln1b = (const float*)d_in[9];
  const float* ln2g = (const float*)d_in[10];
  const float* ln2b = (const float*)d_in[11];
  const float* W1   = (const float*)d_in[12];
  const float* b1   = (const float*)d_in[13];
  const float* W2   = (const float*)d_in[14];
  const float* b2   = (const float*)d_in[15];
  const float* lnfg = (const float*)d_in[16];
  const float* lnfb = (const float*)d_in[17];
  const float* Wout = (const float*)d_in[18];
  const float* bout = (const float*)d_in[19];

  const size_t M = 4096, D = 1024;
  char* ws = (char*)d_ws;
  size_t off = 0;
  float* x   = (float*)(ws + off); off += M * D * 4;            // 16 MB
  u16* lnb   = (u16*)(ws + off);   off += M * D * 2;            //  8 MB
  u16* qb    = (u16*)(ws + off);   off += M * D * 2;            //  8 MB
  u16* kb    = (u16*)(ws + off);   off += M * D * 2;            //  8 MB
  u16* vb    = (u16*)(ws + off);   off += M * D * 2;            //  8 MB
  u16* ab    = (u16*)(ws + off);   off += M * D * 2;            //  8 MB
  u16* h1    = (u16*)(ws + off);   off += M * 4 * D * 2;        // 32 MB
  u16* wb    = (u16*)(ws + off);   off += (size_t)D * 32000 * 2;// 64 MB scratch
  // sub-regions of the weight scratch (used sequentially / disjointly)
  u16* wpkq = wb;                 // 2 MB
  u16* wpkk = wb + D * D;         // 2 MB
  u16* wpkv = wb + 2 * D * D;     // 2 MB
  u16* wob  = wb + 3 * D * D;     // 2 MB
  u16* w1b  = wb + 4 * D * D;     // 8 MB
  u16* w2b  = wb + 8 * D * D;     // 8 MB

  float* out = (float*)d_out;

  embed_kernel<<<16384, 256, 0, stream>>>(tokens, tok_emb, pos_emb, x);

  for (int l = 0; l < 6; l++) {
    // ----- attention block -----
    ln_kernel<<<4096, 256, 0, stream>>>(x, ln1g + l * 1024, ln1b + l * 1024, lnb);
    pack_qkv_kernel<<<4096, 256, 0, stream>>>(Wq + (size_t)l * 16 * 1024 * 64, wpkq);
    pack_qkv_kernel<<<4096, 256, 0, stream>>>(Wk + (size_t)l * 16 * 1024 * 64, wpkk);
    pack_qkv_kernel<<<4096, 256, 0, stream>>>(Wv + (size_t)l * 16 * 1024 * 64, wpkv);
    gemm_wmma<false, false, true><<<dim3(8, 32), 256, 0, stream>>>(
        lnb, wpkq, nullptr, nullptr, qb, 4096, 1024, 1024);
    gemm_wmma<false, false, true><<<dim3(8, 32), 256, 0, stream>>>(
        lnb, wpkk, nullptr, nullptr, kb, 4096, 1024, 1024);
    gemm_wmma<false, false, true><<<dim3(8, 32), 256, 0, stream>>>(
        lnb, wpkv, nullptr, nullptr, vb, 4096, 1024, 1024);
    attn_kernel<<<dim3(8, 64), 256, 0, stream>>>(qb, kb, vb, ab);
    cvt_kernel<<<4096, 256, 0, stream>>>(Wo + (size_t)l * 1024 * 1024, wob);
    gemm_wmma<false, true, false><<<dim3(8, 32), 256, 0, stream>>>(
        ab, wob, bo + l * 1024, x, x, 4096, 1024, 1024);

    // ----- MLP block -----
    ln_kernel<<<4096, 256, 0, stream>>>(x, ln2g + l * 1024, ln2b + l * 1024, lnb);
    cvt_kernel<<<16384, 256, 0, stream>>>(W1 + (size_t)l * 1024 * 4096, w1b);
    gemm_wmma<true, false, true><<<dim3(32, 32), 256, 0, stream>>>(
        lnb, w1b, b1 + l * 4096, nullptr, h1, 4096, 4096, 1024);
    cvt_kernel<<<16384, 256, 0, stream>>>(W2 + (size_t)l * 4096 * 1024, w2b);
    gemm_wmma<false, true, false><<<dim3(8, 32), 256, 0, stream>>>(
        h1, w2b, b2 + l * 1024, x, x, 4096, 1024, 4096);
  }

  // ----- final LN + logits -----
  ln_kernel<<<4096, 256, 0, stream>>>(x, lnfg, lnfb, lnb);
  cvt_kernel<<<128000, 256, 0, stream>>>(Wout, wb);
  gemm_wmma<false, false, false><<<dim3(250, 32), 256, 0, stream>>>(
      lnb, wb, bout, nullptr, out, 4096, 32000, 1024);
}
